// SingleHeadAttention_63024350101722
// MI455X (gfx1250) — matrix operand order repaired and checked by reading, not compiled, not run on hardware
//
#include <hip/hip_runtime.h>
#include <math.h>

#define BB 8
#define CC 2048
#define EE 1024
#define HH 128

typedef __attribute__((ext_vector_type(16))) __bf16 v16bf;
typedef __attribute__((ext_vector_type(8)))  __bf16 v8bf;
typedef __attribute__((ext_vector_type(8)))  float  v8f;
typedef __attribute__((ext_vector_type(4)))  float  v4f;

static __device__ __forceinline__ v16bf join16(v8bf lo, v8bf hi) {
  return __builtin_shufflevector(lo, hi, 0,1,2,3,4,5,6,7,8,9,10,11,12,13,14,15);
}
// load 32 contiguous bytes (16 bf16) as one WMMA operand fragment
static __device__ __forceinline__ v16bf load_frag(const __bf16* p) {
  v8bf lo = *(const v8bf*)p;
  v8bf hi = *(const v8bf*)(p + 8);
  return join16(lo, hi);
}
static __device__ __forceinline__ v8f wmma_bf16(v16bf a, v16bf b, v8f c) {
  // D = A(16x32) * B(32x16) + C, f32 accumulate
  return __builtin_amdgcn_wmma_f32_16x16x32_bf16(false, a, false, b, (short)0, c, false, false);
}
static __device__ __forceinline__ v8f zero8() {
  v8f z;
#pragma unroll
  for (int i = 0; i < 8; ++i) z[i] = 0.0f;
  return z;
}

// ---------------------------------------------------------------------------
// Projection GEMM:  out(row, n) = sum_k x(row, k) * W(k, n)
//   x: (B*C, E) fp32 row-major, W: (E, H) fp32 row-major
//   out: bf16, (B*C, H) when TRANS=0, or transposed (B, H, C) when TRANS=1
// Block = 256 threads = 8 waves; each wave computes a 16-row x 128-col strip.
// W K-chunk (32 x 128) staged in LDS as bf16, transposed to [col][k].
// ---------------------------------------------------------------------------
template <int TRANS>
__global__ __launch_bounds__(256) void proj_gemm_kernel(
    const float* __restrict__ x, const float* __restrict__ W,
    __bf16* __restrict__ out)
{
  __shared__ __align__(16) __bf16 wlds[HH][40];   // [col][k_local], 80B row pitch, 16B-aligned subtiles

  const int tid  = threadIdx.x;
  const int wave = tid >> 5;
  const int lane = tid & 31;
  const int m    = lane & 15;        // A-row / B-col / D-col within tile
  const int hs   = lane >> 4;        // lane-half select

  const int row0 = (blockIdx.x * 8 + wave) * 16;   // 16 output rows per wave
  const float* xrow = x + (size_t)(row0 + m) * EE;

  v8f acc[8];
#pragma unroll
  for (int c = 0; c < 8; ++c) acc[c] = zero8();

  for (int k0 = 0; k0 < EE; k0 += 32) {
    __syncthreads();
    // Cooperative load of W[k0..k0+31][0..127] -> wlds[col][k], bf16
#pragma unroll
    for (int i = 0; i < 16; ++i) {
      int flat = i * 256 + tid;          // 4096 elements
      int kl   = flat >> 7;              // 0..31
      int col  = flat & 127;             // 0..127
      wlds[col][kl] = (__bf16)W[(size_t)(k0 + kl) * HH + col];
    }
    __syncthreads();

    // A fragment: row = row0+m, K = k0 + {hs*8 + e (e<8), 16 + hs*8 + e-8}
    const int g0 = k0 + hs * 8;
    v4f p0 = *(const v4f*)(xrow + g0);
    v4f p1 = *(const v4f*)(xrow + g0 + 4);
    v4f p2 = *(const v4f*)(xrow + g0 + 16);
    v4f p3 = *(const v4f*)(xrow + g0 + 20);
    // prefetch next K-chunk of this row (streams through L2)
    if (k0 + 32 < EE) __builtin_prefetch(xrow + g0 + 32, 0, 1);
    v16bf a;
#pragma unroll
    for (int i = 0; i < 4; ++i) {
      a[i]      = (__bf16)p0[i];
      a[4 + i]  = (__bf16)p1[i];
      a[8 + i]  = (__bf16)p2[i];
      a[12 + i] = (__bf16)p3[i];
    }

    // 8 column tiles of H; B fragment: col = c*16+m, K = {hs*8 + e, 16 + hs*8 + e-8}
    // Load ALL fragments, then fence the scheduler so the loads cannot be
    // sunk down to their consumers (keeps 8 fragments live -> distinct regs,
    // batched ds_loads, staggered dscnt waits).
    v16bf bfrag[8];
#pragma unroll
    for (int c = 0; c < 8; ++c)
      bfrag[c] = join16(*(const v8bf*)&wlds[c * 16 + m][hs * 8], *(const v8bf*)&wlds[c * 16 + m][16 + hs * 8]);
    __builtin_amdgcn_sched_barrier(0);
#pragma unroll
    for (int c = 0; c < 8; ++c)
      acc[c] = wmma_bf16(a, bfrag[c], acc[c]);
  }

  // D layout: row = r + hs*8, col = c*16 + m   (branch-free per instantiation)
  if (TRANS == 0) {
    __bf16* obase = out + (size_t)(row0 + hs * 8) * HH + m;
#pragma unroll
    for (int r = 0; r < 8; ++r) {
      __bf16* orow = obase + (size_t)r * HH;
#pragma unroll
      for (int c = 0; c < 8; ++c) {
        orow[c * 16] = (__bf16)acc[c][r];
      }
    }
  } else {
    const int rowg0 = row0 + hs * 8;
    const int b     = rowg0 >> 11;          // / C (2048); tile never straddles a batch
    const int cc0   = rowg0 & (CC - 1);
    __bf16* obase = out + ((size_t)b * HH + m) * CC + cc0;
#pragma unroll
    for (int c = 0; c < 8; ++c) {
      __bf16* ocol = obase + (size_t)(c * 16) * CC;
#pragma unroll
      for (int r = 0; r < 8; ++r) {
        ocol[r] = (__bf16)acc[c][r];
      }
    }
  }
}

// ---------------------------------------------------------------------------
// Causal flash attention, one 16-row Q tile per wave, Bc = 32 keys per step.
//   qb, kb: (B, C, H) bf16 row-major; vt: (B, H, C) bf16 (V transposed)
//   out: (B, C, H) fp32
// ---------------------------------------------------------------------------
__global__ __launch_bounds__(256) void attn_kernel(
    const __bf16* __restrict__ qb, const __bf16* __restrict__ kb,
    const __bf16* __restrict__ vt, float* __restrict__ out)
{
  __shared__ __align__(16) __bf16 plds[8][16][32];   // per-wave P tile staging

  const int tid  = threadIdx.x;
  const int wave = tid >> 5;
  const int lane = tid & 31;
  const int m    = lane & 15;
  const int hs   = lane >> 4;

  const int bIdx  = blockIdx.x >> 4;          // batch
  const int grp   = blockIdx.x & 15;
  const int i0    = (grp * 8 + wave) * 16;    // first Q row of this wave's tile
  const float scale = 0.03125f;               // 1/sqrt(E) = 1/32

  const __bf16* kbase = kb + (size_t)bIdx * CC * HH;
  const __bf16* vbase = vt + (size_t)bIdx * HH * CC;

  // Q fragments for K-dim = H (4 chunks of 32)
  v16bf qfrag[4];
  {
    const __bf16* qrow = qb + ((size_t)bIdx * CC + i0 + m) * HH;
#pragma unroll
    for (int hc = 0; hc < 4; ++hc) {
      v8bf lo = *(const v8bf*)(qrow + hc * 32 + hs * 8);
      v8bf hi = *(const v8bf*)(qrow + hc * 32 + 16 + hs * 8);
      qfrag[hc] = join16(lo, hi);
    }
  }

  float rmax[8], rsum[8];
  v8f oacc[8];
#pragma unroll
  for (int r = 0; r < 8; ++r) { rmax[r] = -INFINITY; rsum[r] = 0.0f; }
#pragma unroll
  for (int t = 0; t < 8; ++t) oacc[t] = zero8();

  const int jend = i0 + 15;
  for (int j0 = 0; j0 <= jend; j0 += 32) {
    // Prefetch next j-block's K rows (this lane's row) while computing this one.
    if (j0 + 32 <= jend) {
      __builtin_prefetch(kbase + (size_t)(j0 + 32 + m) * HH + hs * 64, 0, 1);
      __builtin_prefetch(vbase + (size_t)m * CC + j0 + 32, 0, 1);
    }

    // ---- S = Q * K^T for 32 keys (two 16-col subtiles) ----
    // K fragment (idx = hc*2 + nt): col = j0 + nt*16 + m, K-dim = hc*32 + {hs*8 + e, 16 + hs*8 + e-8}
    // Issue all 16 global_load_b128, then a sched_barrier stops the machine
    // scheduler from sinking them to the consumers -> staggered loadcnt waits.
    v16bf kfrag[8];
#pragma unroll
    for (int idx = 0; idx < 8; ++idx) {
      int hc = idx >> 1, nt = idx & 1;
      kfrag[idx] = join16(*(const v8bf*)(kbase + (size_t)(j0 + nt * 16 + m) * HH + hc * 32 + hs * 8), *(const v8bf*)(kbase + (size_t)(j0 + nt * 16 + m) * HH + hc * 32 + 16 + hs * 8));
    }
    __builtin_amdgcn_sched_barrier(0);
    v8f sacc[2];
    sacc[0] = zero8();
    sacc[1] = zero8();
#pragma unroll
    for (int idx = 0; idx < 8; ++idx)
      sacc[idx & 1] = wmma_bf16(qfrag[idx >> 1], kfrag[idx], sacc[idx & 1]);

    // ---- causal mask + online softmax (rows r+hs*8, cols j0+m / j0+16+m) ----
#pragma unroll
    for (int r = 0; r < 8; ++r) {
      int row = i0 + r + hs * 8;
      int c0  = j0 + m;
      int c1  = j0 + 16 + m;
      float s0 = sacc[0][r] * scale;
      float s1 = sacc[1][r] * scale;
      if (c0 > row) s0 = -INFINITY;
      if (c1 > row) s1 = -INFINITY;

      float ml = fmaxf(s0, s1);
      ml = fmaxf(ml, __shfl_xor(ml, 1, 32));
      ml = fmaxf(ml, __shfl_xor(ml, 2, 32));
      ml = fmaxf(ml, __shfl_xor(ml, 4, 32));
      ml = fmaxf(ml, __shfl_xor(ml, 8, 32));
      float nm = fmaxf(rmax[r], ml);

      float p0 = __expf(s0 - nm);
      float p1 = __expf(s1 - nm);
      float ps = p0 + p1;
      ps += __shfl_xor(ps, 1, 32);
      ps += __shfl_xor(ps, 2, 32);
      ps += __shfl_xor(ps, 4, 32);
      ps += __shfl_xor(ps, 8, 32);

      float corr = __expf(rmax[r] - nm);
      rsum[r] = rsum[r] * corr + ps;
      rmax[r] = nm;
#pragma unroll
      for (int t = 0; t < 8; ++t) oacc[t][r] *= corr;

      plds[wave][r + hs * 8][m]      = (__bf16)p0;
      plds[wave][r + hs * 8][16 + m] = (__bf16)p1;
    }

    // LDS is in-order per wave; keep compiler from moving the reads up.
    __builtin_amdgcn_wave_barrier();

    // ---- re-read P in A-fragment layout: row = m, K = {hs*8+e, 16+hs*8+e} ----
    const __bf16* prow = &plds[wave][m][0];
    v8bf plo = *(const v8bf*)(prow + hs * 8);
    v8bf phi = *(const v8bf*)(prow + 16 + hs * 8);
    v16bf pfrag = join16(plo, phi);

    __builtin_amdgcn_wave_barrier();

    // ---- O += P * V  (B fragment from transposed V: contiguous in C) ----
    // Same batched-load + sched_barrier scheme for the V fragments.
    v16bf vfrag[8];
#pragma unroll
    for (int t = 0; t < 8; ++t)
      vfrag[t] = join16(*(const v8bf*)(vbase + (size_t)(t * 16 + m) * CC + j0 + hs * 8), *(const v8bf*)(vbase + (size_t)(t * 16 + m) * CC + j0 + 16 + hs * 8));
    __builtin_amdgcn_sched_barrier(0);
#pragma unroll
    for (int t = 0; t < 8; ++t)
      oacc[t] = wmma_bf16(pfrag, vfrag[t], oacc[t]);
  }

  // ---- normalize and write out (fp32), D layout row = r+hs*8, col = t*16+m ----
  float* obase = out + ((size_t)bIdx * CC + i0 + hs * 8) * HH + m;
#pragma unroll
  for (int r = 0; r < 8; ++r) {
    float inv = 1.0f / rsum[r];
    float* orow = obase + (size_t)r * HH;
#pragma unroll
    for (int t = 0; t < 8; ++t) {
      orow[t * 16] = oacc[t][r] * inv;
    }
  }
}

extern "C" void kernel_launch(void* const* d_in, const int* in_sizes, int n_in,
                              void* d_out, int out_size, void* d_ws, size_t ws_size,
                              hipStream_t stream) {
  (void)in_sizes; (void)n_in; (void)out_size; (void)ws_size;
  const float* x  = (const float*)d_in[0];
  const float* Wq = (const float*)d_in[1];
  const float* Wk = (const float*)d_in[2];
  const float* Wv = (const float*)d_in[3];
  float* out = (float*)d_out;

  const size_t nqkv = (size_t)BB * CC * HH;   // 2,097,152 elements each
  __bf16* qb = (__bf16*)d_ws;                 // (B, C, H) bf16
  __bf16* kb = qb + nqkv;                     // (B, C, H) bf16
  __bf16* vt = kb + nqkv;                     // (B, H, C) bf16 (transposed)

  // (B*C)/(16 rows * 8 waves) = 128 blocks
  proj_gemm_kernel<0><<<dim3(128), dim3(256), 0, stream>>>(x, Wq, qb);
  proj_gemm_kernel<0><<<dim3(128), dim3(256), 0, stream>>>(x, Wk, kb);
  proj_gemm_kernel<1><<<dim3(128), dim3(256), 0, stream>>>(x, Wv, vt);

  // 8 batches * 16 tile-groups = 128 blocks, 8 Q-tiles (waves) each
  attn_kernel<<<dim3(128), dim3(256), 0, stream>>>(qb, kb, vt, out);
}
